// MultiHeadSelfAttention_24472723653246
// MI455X (gfx1250) — compile-verified, matrix operand order checked by reading, not run on hardware
//
#include <hip/hip_runtime.h>

// MI455X (gfx1250) multi-head self-attention forward.
// bf16 WMMA (v_wmma_f32_16x16x32_bf16) for QKV GEMM, Q·K^T, P·V, and out-proj.
// GEMM staging uses CDNA5 async global->LDS copies (ASYNCcnt) with LDS
// double-buffering. Softmax in fp32 VALU with wave32 half-lane shuffles.

typedef __bf16 bf16_t;
typedef bf16_t v16bf __attribute__((ext_vector_type(16)));
typedef float  v8f   __attribute__((ext_vector_type(8)));

#define WMMA_BF16(a, b, c) \
  __builtin_amdgcn_wmma_f32_16x16x32_bf16(false, (a), false, (b), (short)0, (c), false, false)

static constexpr int Bb = 4, Tt = 2048, Cc = 2048, Hh = 16, Dd = 128;
static constexpr int Mrows = Bb * Tt;  // 8192

// Load one 16x32 bf16 fragment slice for this lane: 8 bf16 at p, 8 bf16 at p+16.
// Matches the ISA A/B 16-bit fragment layout (lanes 0-15: K 0-7 & 16-23, lanes
// 16-31: K 8-15 & 24-31) when p = rowbase + kh, kh = (lane<16)?0:8.
__device__ __forceinline__ v16bf load_frag(const bf16_t* p) {
  union { v16bf v; uint4 q[2]; } u;
  u.q[0] = *reinterpret_cast<const uint4*>(p);
  u.q[1] = *reinterpret_cast<const uint4*>(p + 16);
  return u.v;
}

// ---------------- fp32 -> bf16 converts ----------------
__global__ void k_cvt(const float* __restrict__ in, bf16_t* __restrict__ out, int n) {
  int i = (blockIdx.x * blockDim.x + threadIdx.x) * 4;
  if (i + 3 < n) {
    float4 f = *reinterpret_cast<const float4*>(in + i);
    union { bf16_t h[4]; uint2 u2; } u;
    u.h[0] = (bf16_t)f.x; u.h[1] = (bf16_t)f.y;
    u.h[2] = (bf16_t)f.z; u.h[3] = (bf16_t)f.w;
    *reinterpret_cast<uint2*>(out + i) = u.u2;
  }
}

// Wt[n][k] = (bf16) W[k][n]   (W row-major [K][N])
__global__ void k_cvt_t(const float* __restrict__ W, bf16_t* __restrict__ Wt,
                        int K, int N) {
  int idx = blockIdx.x * blockDim.x + threadIdx.x;
  if (idx >= K * N) return;
  int k = idx % K, n = idx / K;
  Wt[idx] = (bf16_t)W[(size_t)k * N + n];
}

// ---------------- bf16 WMMA GEMM: C[M,N] = A[M,K] * Bt[N,K]^T ----------------
// Block tile 128x128, 8 waves, wave tile 32x64 (2x4 WMMA 16x16 tiles), K-step 32.
// Staging: GLOBAL_LOAD_ASYNC_TO_LDS_B128 (4 per thread per tile) into a
// double-buffered LDS tile; s_wait_asynccnt 0x4 retires the previous tile's
// in-order async ops while the next tile's 4 remain in flight.
// mode 0: write fp32 row-major to outF.
// mode 1: N = 6144 = [q|k|v] x 16 heads x 128; scatter bf16 to Qb/Kb [B,H,T,D]
//         and V transposed into Vt [B,H,D,T]. The q/k/v select, head, and batch
//         are wave-uniform per 16x16 tile (boundaries are multiples of 16/128),
//         so the epilogue uses scalar branches, and the Vt store packs 8
//         consecutive-t bf16 into one b128 store per lane.
__global__ __launch_bounds__(256)
void k_gemm(const bf16_t* __restrict__ A, const bf16_t* __restrict__ Bt,
            int N, int K, int mode, float* __restrict__ outF,
            bf16_t* __restrict__ Qb, bf16_t* __restrict__ Kb,
            bf16_t* __restrict__ Vt) {
  __shared__ alignas(16) bf16_t lA[2][128 * 32];
  __shared__ alignas(16) bf16_t lB[2][128 * 32];

  const int tid  = threadIdx.x;
  const int lane = tid & 31, wave = tid >> 5;
  const int wm = wave & 3, wn = wave >> 2;
  const int m0 = blockIdx.y * 128, n0 = blockIdx.x * 128;
  const int r  = tid >> 1;              // staging row 0..127
  const int cs = (tid & 1) * 16;        // staging col 0 or 16
  const int rr = lane & 15;
  const int kh = (lane < 16) ? 0 : 8;

  const bf16_t* gArow = A  + (size_t)(m0 + r) * K + cs;
  const bf16_t* gBrow = Bt + (size_t)(n0 + r) * K + cs;

  // Async-stage one 128x32 A tile + B tile (this thread's 2x16B per tile).
  // IOFFSET is added to BOTH the LDS and global address (ISA 15.18.3), so the
  // second 16B chunk is just offset:16 on the same base operands.
  auto stage = [&](int k0, int buf) {
    unsigned la = (unsigned)(uintptr_t)&lA[buf][r * 32 + cs];   // low 32 = LDS off
    unsigned lb = (unsigned)(uintptr_t)&lB[buf][r * 32 + cs];
    unsigned long long ga = (unsigned long long)(uintptr_t)(gArow + k0);
    unsigned long long gb = (unsigned long long)(uintptr_t)(gBrow + k0);
    asm volatile(
        "global_load_async_to_lds_b128 %0, %1, off\n\t"
        "global_load_async_to_lds_b128 %0, %1, off offset:16\n\t"
        "global_load_async_to_lds_b128 %2, %3, off\n\t"
        "global_load_async_to_lds_b128 %2, %3, off offset:16"
        :: "v"(la), "v"(ga), "v"(lb), "v"(gb) : "memory");
  };

  v8f acc[2][4] = {};

  stage(0, 0);
  int buf = 0;
  for (int k0 = 0; k0 < K; k0 += 32, buf ^= 1) {
    // Prefetch next tile (wraps to tile 0 on the last iter so the asynccnt
    // wait value stays a uniform immediate; the extra tile is never read).
    int kn = (k0 + 32 < K) ? (k0 + 32) : 0;
    stage(kn, buf ^ 1);
    // Previous 4 async ops (current tile) done; next 4 may stay in flight.
    asm volatile("s_wait_asynccnt 0x4" ::: "memory");
    __syncthreads();

    v16bf a0 = load_frag(&lA[buf][(wm * 32 + rr) * 32 + kh]);
    v16bf a1 = load_frag(&lA[buf][(wm * 32 + 16 + rr) * 32 + kh]);
    v16bf bfr[4];
#pragma unroll
    for (int nt = 0; nt < 4; ++nt)
      bfr[nt] = load_frag(&lB[buf][(wn * 64 + nt * 16 + rr) * 32 + kh]);
#pragma unroll
    for (int nt = 0; nt < 4; ++nt) {
      acc[0][nt] = WMMA_BF16(a0, bfr[nt], acc[0][nt]);
      acc[1][nt] = WMMA_BF16(a1, bfr[nt], acc[1][nt]);
    }
    __syncthreads();
  }

  const int rowoff = (lane < 16) ? 0 : 8;
#pragma unroll
  for (int mt = 0; mt < 2; ++mt) {
    const int rowb = m0 + wm * 32 + mt * 16 + rowoff;   // + j per element
#pragma unroll
    for (int nt = 0; nt < 4; ++nt) {
      const int colb = n0 + wn * 64 + nt * 16;          // wave-uniform
      const int col  = colb + rr;
      if (mode == 0) {
#pragma unroll
        for (int j = 0; j < 8; ++j)
          outF[(size_t)(rowb + j) * N + col] = acc[mt][nt][j];
      } else {
        const int s    = colb >> 11;          // uniform: q/k/v select
        const int remb = colb & 2047;
        const int h    = remb >> 7;           // uniform: head
        const int d    = (remb & 127) + rr;   // per-lane d within head
        const int b    = rowb >> 11;          // uniform: batch (T rows per b)
        const int tb   = rowb & 2047;         // base t; + j per element
        const size_t hd = (size_t)(b * Hh + h);
        if (s == 0) {
          bf16_t* dst = Qb + (hd * Tt + tb) * Dd + d;
#pragma unroll
          for (int j = 0; j < 8; ++j) dst[(size_t)j * Dd] = (bf16_t)acc[mt][nt][j];
        } else if (s == 1) {
          bf16_t* dst = Kb + (hd * Tt + tb) * Dd + d;
#pragma unroll
          for (int j = 0; j < 8; ++j) dst[(size_t)j * Dd] = (bf16_t)acc[mt][nt][j];
        } else {
          // Vt[(hd*Dd + d)*Tt + tb + j]: 8 consecutive bf16 -> one b128 store
          union { bf16_t hv[8]; uint4 q; } u;
#pragma unroll
          for (int j = 0; j < 8; ++j) u.hv[j] = (bf16_t)acc[mt][nt][j];
          *reinterpret_cast<uint4*>(Vt + (hd * Dd + d) * Tt + tb) = u.q;
        }
      }
    }
  }
}

// ---------------- flash attention ----------------
// Grid: B*H*(T/64) blocks of 128 threads (4 waves). Wave w owns q rows
// [q0, q0+16). Online softmax in fp32; per-wave LDS round-trip converts the
// P score fragment (C layout) into an A fragment for the P*V WMMAs.
__global__ __launch_bounds__(128)
void k_attn(const bf16_t* __restrict__ Qb, const bf16_t* __restrict__ Kb,
            const bf16_t* __restrict__ Vt, const unsigned char* __restrict__ tok,
            bf16_t* __restrict__ Yb) {
  __shared__ alignas(16) bf16_t lP[4][16 * 32];

  const int tid = threadIdx.x, lane = tid & 31, wave = tid >> 5;
  const int nqb = Tt / 64;                         // 32 q-blocks per (b,h)
  const int qb = blockIdx.x % nqb;
  const int h  = (blockIdx.x / nqb) % Hh;
  const int b  = blockIdx.x / (nqb * Hh);
  const int q0 = qb * 64 + wave * 16;
  const int rr = lane & 15;
  const int kh = (lane < 16) ? 0 : 8;
  const int rowoff = (lane < 16) ? 0 : 8;

  const bf16_t* Qh = Qb + (size_t)(b * Hh + h) * Tt * Dd;
  const bf16_t* Kh = Kb + (size_t)(b * Hh + h) * Tt * Dd;
  const bf16_t* Vh = Vt + (size_t)(b * Hh + h) * Dd * Tt;
  const unsigned char* tm = tok + (size_t)b * Tt;  // jnp bool -> 1 byte/elem

  v16bf qf[4];
#pragma unroll
  for (int db = 0; db < 4; ++db)
    qf[db] = load_frag(Qh + (size_t)(q0 + rr) * Dd + db * 32 + kh);

  v8f o[8] = {};
  float mrow[8], lrow[8];
#pragma unroll
  for (int j = 0; j < 8; ++j) { mrow[j] = -1e30f; lrow[j] = 0.f; }

  const float scale = 0.08838834764831845f;        // 1/sqrt(128)
  const int kend = q0 + 16;                        // exclusive causal bound

  for (int k0 = 0; k0 < kend; k0 += 32) {
    // ---- scores: S(16q x 32k) via two 16x16 WMMA column tiles over D=128
    v8f s0 = {}, s1 = {};
#pragma unroll
    for (int db = 0; db < 4; ++db) {
      v16bf kf0 = load_frag(Kh + (size_t)(k0 + rr) * Dd + db * 32 + kh);
      v16bf kf1 = load_frag(Kh + (size_t)(k0 + 16 + rr) * Dd + db * 32 + kh);
      s0 = WMMA_BF16(qf[db], kf0, s0);
      s1 = WMMA_BF16(qf[db], kf1, s1);
    }
    const int key0 = k0 + rr, key1 = key0 + 16;
    const bool tv0 = tm[key0] != 0, tv1 = tm[key1] != 0;

    float e0[8], e1[8], mx[8];
    int okm = 0;
#pragma unroll
    for (int j = 0; j < 8; ++j) {
      int qrow = q0 + j + rowoff;
      bool o0 = tv0 && (key0 <= qrow);
      bool o1 = tv1 && (key1 <= qrow);
      e0[j] = o0 ? s0[j] * scale : -1e30f;
      e1[j] = o1 ? s1[j] * scale : -1e30f;
      okm |= (o0 ? 1 : 0) << j;
      okm |= (o1 ? 1 : 0) << (j + 8);
      mx[j] = fmaxf(e0[j], e1[j]);
    }
    // row max over the 16 lanes of each half (xor 1,2,4,8 stays in-half)
#pragma unroll
    for (int j = 0; j < 8; ++j) {
      float v = mx[j];
      v = fmaxf(v, __shfl_xor(v, 1, 32));
      v = fmaxf(v, __shfl_xor(v, 2, 32));
      v = fmaxf(v, __shfl_xor(v, 4, 32));
      v = fmaxf(v, __shfl_xor(v, 8, 32));
      mx[j] = v;
    }
#pragma unroll
    for (int j = 0; j < 8; ++j) {
      float mnew = fmaxf(mrow[j], mx[j]);
      float corr = __expf(mrow[j] - mnew);
      mrow[j] = mnew;
      float p0 = ((okm >> j) & 1)       ? __expf(e0[j] - mnew) : 0.f;
      float p1 = ((okm >> (j + 8)) & 1) ? __expf(e1[j] - mnew) : 0.f;
      lrow[j] = lrow[j] * corr + p0 + p1;         // per-lane partial row sum
#pragma unroll
      for (int dt = 0; dt < 8; ++dt) o[dt][j] *= corr;
      lP[wave][(j + rowoff) * 32 + rr]      = (bf16_t)p0;
      lP[wave][(j + rowoff) * 32 + rr + 16] = (bf16_t)p1;
    }
    // same-wave LDS ops are in-order (DScnt in-order); just stop the compiler
    // from hoisting the reload above the stores.
    asm volatile("" ::: "memory");
    v16bf pf = load_frag(&lP[wave][rr * 32 + kh]);
    // ---- O(16q x 128d) += P(16x32) * V(32x128); Vt rows are d, cols keys
#pragma unroll
    for (int dt = 0; dt < 8; ++dt) {
      v16bf vf = load_frag(Vh + (size_t)(dt * 16 + rr) * Tt + k0 + kh);
      o[dt] = WMMA_BF16(pf, vf, o[dt]);
    }
  }

  // final row sums across the 16-lane halves, then normalize + store
#pragma unroll
  for (int j = 0; j < 8; ++j) {
    float v = lrow[j];
    v += __shfl_xor(v, 1, 32);
    v += __shfl_xor(v, 2, 32);
    v += __shfl_xor(v, 4, 32);
    v += __shfl_xor(v, 8, 32);
    lrow[j] = 1.0f / v;                 // row 0 always valid -> v > 0
  }
#pragma unroll
  for (int dt = 0; dt < 8; ++dt)
#pragma unroll
    for (int j = 0; j < 8; ++j) {
      int q = q0 + j + rowoff;
      Yb[(size_t)(b * Tt + q) * Cc + h * Dd + dt * 16 + rr] =
          (bf16_t)(o[dt][j] * lrow[j]);
    }
}

// ---------------- host launcher ----------------
extern "C" void kernel_launch(void* const* d_in, const int* in_sizes, int n_in,
                              void* d_out, int out_size, void* d_ws, size_t ws_size,
                              hipStream_t stream) {
  (void)in_sizes; (void)n_in; (void)out_size; (void)ws_size;
  const float* x     = (const float*)d_in[0];
  const unsigned char* tok = (const unsigned char*)d_in[1];  // bool mask, 1B/elem
  const float* Wqkv  = (const float*)d_in[2];
  const float* Wproj = (const float*)d_in[3];
  float* out = (float*)d_out;

  // workspace layout (bf16 elements), total ~192 MB
  bf16_t* p = (bf16_t*)d_ws;
  bf16_t* xb     = p; p += (size_t)Mrows * Cc;        // x in bf16
  bf16_t* wqkvt  = p; p += (size_t)3 * Cc * Cc;       // W_qkv^T [6144][2048]
  bf16_t* wprojt = p; p += (size_t)Cc * Cc;           // W_proj^T [2048][2048]
  bf16_t* Qb     = p; p += (size_t)Bb * Hh * Tt * Dd; // [B,H,T,D]
  bf16_t* Kb     = p; p += (size_t)Bb * Hh * Tt * Dd; // [B,H,T,D]
  bf16_t* Vt     = p; p += (size_t)Bb * Hh * Tt * Dd; // [B,H,D,T]
  bf16_t* Yb     = p; p += (size_t)Mrows * Cc;        // attn output [B,T,C]

  const int nx = Mrows * Cc;
  k_cvt<<<nx / (256 * 4), 256, 0, stream>>>(x, xb, nx);
  k_cvt_t<<<(3 * Cc * Cc) / 256, 256, 0, stream>>>(Wqkv, wqkvt, Cc, 3 * Cc);
  k_cvt_t<<<(Cc * Cc) / 256, 256, 0, stream>>>(Wproj, wprojt, Cc, Cc);

  dim3 g_qkv(3 * Cc / 128, Mrows / 128);   // 48 x 64
  k_gemm<<<g_qkv, 256, 0, stream>>>(xb, wqkvt, 3 * Cc, Cc, 1,
                                    nullptr, Qb, Kb, Vt);

  k_attn<<<Bb * Hh * (Tt / 64), 128, 0, stream>>>(Qb, Kb, Vt, tok, Yb);

  dim3 g_proj(Cc / 128, Mrows / 128);      // 16 x 64
  k_gemm<<<g_proj, 256, 0, stream>>>(Yb, wprojt, Cc, Cc, 0,
                                     out, nullptr, nullptr, nullptr);
}
